// DynamicTransformerBlock_73864847556959
// MI455X (gfx1250) — compile-verified, matrix operand order checked by reading, not run on hardware
//
#include <hip/hip_runtime.h>
#include <hip/hip_bf16.h>

// ---------------- problem constants (match reference) ----------------
constexpr int B_   = 2;
constexpr int S_   = 2048;
constexpr int DIM_ = 2048;
constexpr int H_   = 16;
constexpr int HKV_ = 4;
constexpr int HD_  = 128;      // DIM/H
constexpr int NREP_= 4;        // H/HKV
constexpr int FF_  = 8192;     // 4*DIM
constexpr float EPS_ = 1e-5f;
constexpr int MROWS = B_ * S_; // 4096

typedef __attribute__((ext_vector_type(16))) _Float16 v16h;
typedef __attribute__((ext_vector_type(8)))  float    v8f;
typedef unsigned int u32x4 __attribute__((ext_vector_type(4)));
typedef int          i32x4 __attribute__((ext_vector_type(4)));
typedef int          i32x8 __attribute__((ext_vector_type(8)));

#if defined(__has_builtin)
#if __has_builtin(__builtin_amdgcn_tensor_load_to_lds) && \
    __has_builtin(__builtin_amdgcn_s_wait_tensorcnt)
#define HAVE_TDM 1
#endif
#endif
#ifndef HAVE_TDM
#define HAVE_TDM 0
#endif

static __device__ __forceinline__ v8f vzero8() {
    v8f z;
    #pragma unroll
    for (int i = 0; i < 8; ++i) z[i] = 0.0f;
    return z;
}

// Load one 16x16x32 f16 fragment half-pair: 8 halves at p0, 8 at p1 (16B aligned).
static __device__ __forceinline__ v16h ldfrag(const _Float16* p0, const _Float16* p1) {
    union { uint4 u[2]; v16h h; } t;
    t.u[0] = *reinterpret_cast<const uint4*>(p0);
    t.u[1] = *reinterpret_cast<const uint4*>(p1);
    return t.h;
}

static __device__ __forceinline__ v8f wmma_f16(v16h a, v16h b, v8f c) {
    return __builtin_amdgcn_wmma_f32_16x16x32_f16(
        false, a, false, b, (short)0, c, false, false);
}

// Issue a TDM 2D tile load: tile_d0 contiguous f16 elems x tile_d1 rows,
// row stride = rowstride (elements), global src -> LDS dst.
static __device__ __forceinline__ void tdm_load_tile_f16(
    const _Float16* gsrc, _Float16* lds_dst,
    int tile_d0, int tile_d1, int rowstride) {
#if HAVE_TDM
    const unsigned long long ga = (unsigned long long)(uintptr_t)gsrc;
    const unsigned int lds_addr = (unsigned int)(uintptr_t)lds_dst;
    u32x4 g0;
    g0[0] = 1u;                                   // count=1, user mode
    g0[1] = lds_addr;                             // lds_addr[31:0]
    g0[2] = (unsigned int)(ga & 0xffffffffu);     // global_addr[31:0]
    g0[3] = (unsigned int)((ga >> 32) & 0x01ffffffu) | 0x80000000u; // addr[56:32] | type=2
    i32x8 g1;
    g1[0] = (int)(1u << 16);                      // data_size=1 (2B), mask=0, no flags
    g1[1] = (int)((0x100000u & 0xffffu) << 16);   // tensor_dim0 lo16 in [31:16]
    g1[2] = (int)((0x100000u >> 16) |             // tensor_dim0 hi16
                  ((0x100000u & 0xffffu) << 16)); // tensor_dim1 lo16
    g1[3] = (int)((0x100000u >> 16) |             // tensor_dim1 hi16
                  ((unsigned int)tile_d0 << 16)); // tile_dim0
    g1[4] = (int)((unsigned int)tile_d1);         // tile_dim1 (tile_dim2=0)
    g1[5] = (int)(unsigned int)rowstride;         // tensor_dim0_stride[31:0]
    g1[6] = 0;                                    // stride hi, dim1_stride lo
    g1[7] = 0;
    i32x4 gz;
    gz[0] = 0; gz[1] = 0; gz[2] = 0; gz[3] = 0;
#if defined(__clang_major__) && (__clang_major__ >= 23)
    i32x8 g1z;
    #pragma unroll
    for (int i = 0; i < 8; ++i) g1z[i] = 0;
    __builtin_amdgcn_tensor_load_to_lds(g0, g1, gz, gz, g1z, 0);
#else
    __builtin_amdgcn_tensor_load_to_lds(g0, g1, gz, gz, 0);
#endif
#else
    (void)gsrc; (void)lds_dst; (void)tile_d0; (void)tile_d1; (void)rowstride;
#endif
}

// ---------------- f32 -> f16 conversion ----------------
__global__ void cvt_f32_f16_kernel(const float* __restrict__ src,
                                   _Float16* __restrict__ dst, int n) {
    int i = blockIdx.x * blockDim.x + threadIdx.x;
    if (i < n) dst[i] = (_Float16)src[i];
}

// ---------------- layernorm (one block per row) ----------------
__global__ void layernorm_f16_kernel(const float* __restrict__ x,
                                     const float* __restrict__ w,
                                     const float* __restrict__ b,
                                     _Float16* __restrict__ out) {
    __shared__ float red[256];
    const int row = blockIdx.x;
    const float* xr = x + (size_t)row * DIM_;
    float s = 0.f;
    for (int i = threadIdx.x; i < DIM_; i += 256) s += xr[i];
    red[threadIdx.x] = s; __syncthreads();
    for (int st = 128; st > 0; st >>= 1) {
        if (threadIdx.x < st) red[threadIdx.x] += red[threadIdx.x + st];
        __syncthreads();
    }
    const float mean = red[0] / (float)DIM_;
    __syncthreads();
    float sq = 0.f;
    for (int i = threadIdx.x; i < DIM_; i += 256) {
        float d = xr[i] - mean; sq += d * d;
    }
    red[threadIdx.x] = sq; __syncthreads();
    for (int st = 128; st > 0; st >>= 1) {
        if (threadIdx.x < st) red[threadIdx.x] += red[threadIdx.x + st];
        __syncthreads();
    }
    const float rstd = rsqrtf(red[0] / (float)DIM_ + EPS_);
    _Float16* o = out + (size_t)row * DIM_;
    for (int i = threadIdx.x; i < DIM_; i += 256)
        o[i] = (_Float16)((xr[i] - mean) * rstd * w[i] + b[i]);
}

// ---------------- RoPE + f16 convert ----------------
__global__ void rope_cvt_kernel(const float* __restrict__ src,
                                _Float16* __restrict__ dst,
                                const float* __restrict__ cosT,
                                const float* __restrict__ sinT,
                                int nh, int dorope) {
    const int total = B_ * S_ * HD_ * nh;
    int i = blockIdx.x * blockDim.x + threadIdx.x;
    if (i >= total) return;
    const int d = i % HD_;
    const int s = (i / (HD_ * nh)) % S_;
    float v = src[i];
    if (dorope) {
        const float c = cosT[s * HD_ + d];
        const float sn = sinT[s * HD_ + d];
        float rot = (d < HD_ / 2) ? -src[i + HD_ / 2] : src[i - HD_ / 2];
        v = v * c + rot * sn;
    }
    dst[i] = (_Float16)v;
}

// ---------------- WMMA GEMM, templated epilogue ----------------
enum { EPI_F32 = 0, EPI_ADD_F32 = 2, EPI_SILU_MUL_F16 = 3 };

// C[M,N] = A_f16[M,K] @ W_f16[N,K]^T. Block 128x128, 8 waves, wave tile 32x64.
// Double-buffered LDS; B panel staged by TDM (wave 0), A panel by vector loads.
template <int EPI>
__global__ void __launch_bounds__(256)
gemm_wmma_kernel(const _Float16* __restrict__ A,
                 const _Float16* __restrict__ W,
                 float* __restrict__ Cf32,
                 _Float16* __restrict__ Cf16,
                 const float* __restrict__ extra,
                 int M, int N, int K) {
    __shared__ _Float16 As[2][128 * 32];
    __shared__ _Float16 Bs[2][128 * 32];

    const int tid  = threadIdx.x;
    const int lane = tid & 31;
    const int wv   = tid >> 5;
    const int wm   = wv & 3;
    const int wn   = wv >> 2;
    const int g    = lane >> 4;
    const int l    = lane & 15;
    const int n0 = blockIdx.x * 128;
    const int m0 = blockIdx.y * 128;

    auto stageA = [&](int buf, int k0) {
        #pragma unroll
        for (int i = 0; i < 2; ++i) {
            const int idx = tid * 2 + i;
            const int r = idx >> 2;
            const int c = (idx & 3) * 8;
            *reinterpret_cast<uint4*>(&As[buf][r * 32 + c]) =
                *reinterpret_cast<const uint4*>(&A[(size_t)(m0 + r) * K + k0 + c]);
        }
    };
    auto stageB = [&](int buf, int k0) {
#if HAVE_TDM
        if (wv == 0)
            tdm_load_tile_f16(&W[(size_t)n0 * K + k0], &Bs[buf][0], 32, 128, K);
#else
        #pragma unroll
        for (int i = 0; i < 2; ++i) {
            const int idx = tid * 2 + i;
            const int r = idx >> 2;
            const int c = (idx & 3) * 8;
            *reinterpret_cast<uint4*>(&Bs[buf][r * 32 + c]) =
                *reinterpret_cast<const uint4*>(&W[(size_t)(n0 + r) * K + k0 + c]);
        }
#endif
    };

    v8f acc[2][4];
    #pragma unroll
    for (int i = 0; i < 2; ++i)
        #pragma unroll
        for (int j = 0; j < 4; ++j) acc[i][j] = vzero8();

    // prologue: fill buffer 0
    stageA(0, 0);
    stageB(0, 0);
#if HAVE_TDM
    if (wv == 0) __builtin_amdgcn_s_wait_tensorcnt(0);
#endif
    __syncthreads();

    const int nsteps = K / 32;
    for (int t = 0; t < nsteps; ++t) {
        const int cur = t & 1;
        if (t + 1 < nsteps) {
            stageA(1 - cur, (t + 1) * 32);
            stageB(1 - cur, (t + 1) * 32);
        }
        v16h aF[2], bF[4];
        #pragma unroll
        for (int sm = 0; sm < 2; ++sm) {
            const _Float16* p = &As[cur][(wm * 32 + sm * 16 + l) * 32];
            aF[sm] = ldfrag(p + g * 8, p + 16 + g * 8);
        }
        #pragma unroll
        for (int sn = 0; sn < 4; ++sn) {
            const _Float16* p = &Bs[cur][(wn * 64 + sn * 16 + l) * 32];
            bF[sn] = ldfrag(p + g * 8, p + 16 + g * 8);
        }
        #pragma unroll
        for (int sm = 0; sm < 2; ++sm)
            #pragma unroll
            for (int sn = 0; sn < 4; ++sn)
                acc[sm][sn] = wmma_f16(aF[sm], bF[sn], acc[sm][sn]);
#if HAVE_TDM
        if (wv == 0 && t + 1 < nsteps) __builtin_amdgcn_s_wait_tensorcnt(0);
#endif
        __syncthreads();
    }

    // straight-line epilogue (compile-time EPI)
    #pragma unroll
    for (int sm = 0; sm < 2; ++sm) {
        #pragma unroll
        for (int sn = 0; sn < 4; ++sn) {
            #pragma unroll
            for (int r = 0; r < 8; ++r) {
                const int row = m0 + wm * 32 + sm * 16 + r + 8 * g;
                const int col = n0 + wn * 64 + sn * 16 + l;
                const size_t idx = (size_t)row * N + col;
                const float v = acc[sm][sn][r];
                if (EPI == EPI_F32) {
                    Cf32[idx] = v;
                } else if (EPI == EPI_ADD_F32) {
                    Cf32[idx] = v + extra[idx];
                } else { // EPI_SILU_MUL_F16
                    const float gv = extra[idx];
                    const float sg = gv / (1.0f + __expf(-gv));
                    Cf16[idx] = (_Float16)(sg * v);
                }
            }
        }
    }
}

// ---------------- flash attention (non-causal, GQA) ----------------
__global__ void __launch_bounds__(256)
flash_attn_kernel(const _Float16* __restrict__ Qh,
                  const _Float16* __restrict__ Kh,
                  const _Float16* __restrict__ Vh,
                  _Float16* __restrict__ Oh) {
    __shared__ _Float16 Kt[32 * 128];
    __shared__ _Float16 VtT[128 * 32];
    __shared__ _Float16 Pst[8 * 16 * 32];

    const int tid  = threadIdx.x;
    const int lane = tid & 31;
    const int wv   = tid >> 5;
    const int g    = lane >> 4;
    const int l    = lane & 15;

    const int qblk = blockIdx.x;
    const int h    = blockIdx.y;
    const int b    = blockIdx.z;
    const int hkv  = h / NREP_;
    const int qbase = qblk * 128 + wv * 16;
    const float scale = 0.08838834764831845f; // 1/sqrt(128)

    v16h aQ[4];
    {
        const int srow = qbase + l;
        const _Float16* qp = Qh + (((size_t)(b * S_ + srow) * H_) + h) * HD_;
        #pragma unroll
        for (int kc = 0; kc < 4; ++kc) {
            const int d0 = kc * 32 + g * 8;
            aQ[kc] = ldfrag(qp + d0, qp + d0 + 16);
        }
    }

    v8f oacc[8];
    #pragma unroll
    for (int t = 0; t < 8; ++t) oacc[t] = vzero8();
    float m_i[8], l_i[8];
    #pragma unroll
    for (int r = 0; r < 8; ++r) { m_i[r] = -1e30f; l_i[r] = 0.0f; }

    const int nkv = S_ / 32;
    for (int kt = 0; kt < nkv; ++kt) {
        __syncthreads();
        #pragma unroll
        for (int i = 0; i < 2; ++i) {
            const int idx = tid * 2 + i;
            const int row = idx >> 4;
            const int c8  = (idx & 15) * 8;
            const size_t gidx = (((size_t)(b * S_ + kt * 32 + row) * HKV_) + hkv) * HD_ + c8;
            *reinterpret_cast<uint4*>(&Kt[row * 128 + c8]) =
                *reinterpret_cast<const uint4*>(&Kh[gidx]);
            union { uint4 u; _Float16 hx[8]; } tv;
            tv.u = *reinterpret_cast<const uint4*>(&Vh[gidx]);
            #pragma unroll
            for (int j = 0; j < 8; ++j) VtT[(c8 + j) * 32 + row] = tv.hx[j];
            if (kt + 1 < nkv) {
                const size_t gnext = gidx + (size_t)32 * HKV_ * HD_;
                __builtin_prefetch(&Kh[gnext], 0, 0);
                __builtin_prefetch(&Vh[gnext], 0, 0);
            }
        }
        __syncthreads();

        v8f sc[2];
        #pragma unroll
        for (int nt = 0; nt < 2; ++nt) {
            sc[nt] = vzero8();
            const _Float16* kp = &Kt[(nt * 16 + l) * 128];
            #pragma unroll
            for (int kc = 0; kc < 4; ++kc) {
                v16h bK = ldfrag(kp + kc * 32 + g * 8, kp + kc * 32 + 16 + g * 8);
                sc[nt] = wmma_f16(aQ[kc], bK, sc[nt]);
            }
            #pragma unroll
            for (int r = 0; r < 8; ++r) sc[nt][r] *= scale;
        }

        #pragma unroll
        for (int r = 0; r < 8; ++r) {
            float mx = fmaxf(sc[0][r], sc[1][r]);
            #pragma unroll
            for (int d = 1; d < 16; d <<= 1)
                mx = fmaxf(mx, __shfl_xor(mx, d, 32));
            const float m_new = fmaxf(m_i[r], mx);
            const float cf = __expf(m_i[r] - m_new);
            const float p0 = __expf(sc[0][r] - m_new);
            const float p1 = __expf(sc[1][r] - m_new);
            float rs = p0 + p1;
            #pragma unroll
            for (int d = 1; d < 16; d <<= 1)
                rs += __shfl_xor(rs, d, 32);
            l_i[r] = l_i[r] * cf + rs;
            m_i[r] = m_new;
            #pragma unroll
            for (int t = 0; t < 8; ++t) oacc[t][r] *= cf;
            const int mrow = r + 8 * g;
            Pst[wv * 512 + mrow * 32 + 0 * 16 + l] = (_Float16)p0;
            Pst[wv * 512 + mrow * 32 + 1 * 16 + l] = (_Float16)p1;
        }

        const _Float16* pp = &Pst[wv * 512 + l * 32];
        v16h aP = ldfrag(pp + g * 8, pp + 16 + g * 8);
        #pragma unroll
        for (int dt = 0; dt < 8; ++dt) {
            const _Float16* vp = &VtT[(dt * 16 + l) * 32];
            v16h bV = ldfrag(vp + g * 8, vp + 16 + g * 8);
            oacc[dt] = wmma_f16(aP, bV, oacc[dt]);
        }
        __syncthreads();
    }

    #pragma unroll
    for (int dt = 0; dt < 8; ++dt) {
        #pragma unroll
        for (int r = 0; r < 8; ++r) {
            const int srow = qbase + r + 8 * g;
            const int d = dt * 16 + l;
            const float v = oacc[dt][r] * (1.0f / l_i[r]);
            Oh[(((size_t)(b * S_ + srow) * H_) + h) * HD_ + d] = (_Float16)v;
        }
    }
}

// ---------------- host orchestration ----------------
extern "C" void kernel_launch(void* const* d_in, const int* in_sizes, int n_in,
                              void* d_out, int out_size, void* d_ws, size_t ws_size,
                              hipStream_t stream) {
    const float* x      = (const float*)d_in[0];
    const float* cosT   = (const float*)d_in[1];
    const float* sinT   = (const float*)d_in[2];
    const float* w_q    = (const float*)d_in[3];
    const float* w_k    = (const float*)d_in[4];
    const float* w_v    = (const float*)d_in[5];
    const float* w_o    = (const float*)d_in[6];
    const float* w_gate = (const float*)d_in[7];
    const float* w_up   = (const float*)d_in[8];
    const float* w_down = (const float*)d_in[9];
    const float* ln1_w  = (const float*)d_in[10];
    const float* ln1_b  = (const float*)d_in[11];
    const float* ln2_w  = (const float*)d_in[12];
    const float* ln2_b  = (const float*)d_in[13];
    float* out = (float*)d_out;

    char* ws = (char*)d_ws;
    size_t off = 0;
    auto alloc = [&](size_t bytes) -> char* {
        char* p = ws + off;
        off = (off + bytes + 255) & ~(size_t)255;
        return p;
    };
    _Float16* wq_h = (_Float16*)alloc((size_t)DIM_ * DIM_ * 2);
    _Float16* wk_h = (_Float16*)alloc((size_t)HKV_ * HD_ * DIM_ * 2);
    _Float16* wv_h = (_Float16*)alloc((size_t)HKV_ * HD_ * DIM_ * 2);
    _Float16* wo_h = (_Float16*)alloc((size_t)DIM_ * DIM_ * 2);
    _Float16* wg_h = (_Float16*)alloc((size_t)FF_ * DIM_ * 2);
    _Float16* wu_h = (_Float16*)alloc((size_t)FF_ * DIM_ * 2);
    _Float16* wd_h = (_Float16*)alloc((size_t)DIM_ * FF_ * 2);
    _Float16* h1   = (_Float16*)alloc((size_t)MROWS * DIM_ * 2);
    float*    qf   = (float*)   alloc((size_t)MROWS * DIM_ * 4);
    float*    kf   = (float*)   alloc((size_t)MROWS * HKV_ * HD_ * 4);
    float*    vf   = (float*)   alloc((size_t)MROWS * HKV_ * HD_ * 4);
    _Float16* qh   = (_Float16*)alloc((size_t)MROWS * DIM_ * 2);
    _Float16* kh   = (_Float16*)alloc((size_t)MROWS * HKV_ * HD_ * 2);
    _Float16* vh   = (_Float16*)alloc((size_t)MROWS * HKV_ * HD_ * 2);
    _Float16* ao   = (_Float16*)alloc((size_t)MROWS * DIM_ * 2);
    float*    x2   = (float*)   alloc((size_t)MROWS * DIM_ * 4);
    _Float16* h2   = (_Float16*)alloc((size_t)MROWS * DIM_ * 2);
    float*    gf   = (float*)   alloc((size_t)MROWS * FF_ * 4);
    _Float16* af   = (_Float16*)alloc((size_t)MROWS * FF_ * 2);
    (void)ws_size; (void)n_in; (void)in_sizes; (void)out_size;

    auto cvt = [&](const float* s, _Float16* d, size_t n) {
        cvt_f32_f16_kernel<<<(int)((n + 255) / 256), 256, 0, stream>>>(s, d, (int)n);
    };
    cvt(w_q,    wq_h, (size_t)DIM_ * DIM_);
    cvt(w_k,    wk_h, (size_t)HKV_ * HD_ * DIM_);
    cvt(w_v,    wv_h, (size_t)HKV_ * HD_ * DIM_);
    cvt(w_o,    wo_h, (size_t)DIM_ * DIM_);
    cvt(w_gate, wg_h, (size_t)FF_ * DIM_);
    cvt(w_up,   wu_h, (size_t)FF_ * DIM_);
    cvt(w_down, wd_h, (size_t)DIM_ * FF_);

    layernorm_f16_kernel<<<MROWS, 256, 0, stream>>>(x, ln1_w, ln1_b, h1);

    {
        dim3 gq(DIM_ / 128, MROWS / 128);
        gemm_wmma_kernel<EPI_F32><<<gq, 256, 0, stream>>>(
            h1, wq_h, qf, nullptr, nullptr, MROWS, DIM_, DIM_);
        dim3 gk((HKV_ * HD_) / 128, MROWS / 128);
        gemm_wmma_kernel<EPI_F32><<<gk, 256, 0, stream>>>(
            h1, wk_h, kf, nullptr, nullptr, MROWS, HKV_ * HD_, DIM_);
        gemm_wmma_kernel<EPI_F32><<<gk, 256, 0, stream>>>(
            h1, wv_h, vf, nullptr, nullptr, MROWS, HKV_ * HD_, DIM_);
    }

    {
        const int nq = MROWS * DIM_;
        rope_cvt_kernel<<<(nq + 255) / 256, 256, 0, stream>>>(qf, qh, cosT, sinT, H_, 1);
        const int nk = MROWS * HKV_ * HD_;
        rope_cvt_kernel<<<(nk + 255) / 256, 256, 0, stream>>>(kf, kh, cosT, sinT, HKV_, 1);
        rope_cvt_kernel<<<(nk + 255) / 256, 256, 0, stream>>>(vf, vh, cosT, sinT, HKV_, 0);
    }

    {
        dim3 ga(S_ / 128, H_, B_);
        flash_attn_kernel<<<ga, 256, 0, stream>>>(qh, kh, vh, ao);
    }

    {
        dim3 go(DIM_ / 128, MROWS / 128);
        gemm_wmma_kernel<EPI_ADD_F32><<<go, 256, 0, stream>>>(
            ao, wo_h, x2, nullptr, x, MROWS, DIM_, DIM_);
    }

    layernorm_f16_kernel<<<MROWS, 256, 0, stream>>>(x2, ln2_w, ln2_b, h2);

    {
        dim3 gg(FF_ / 128, MROWS / 128);
        gemm_wmma_kernel<EPI_F32><<<gg, 256, 0, stream>>>(
            h2, wg_h, gf, nullptr, nullptr, MROWS, FF_, DIM_);
        gemm_wmma_kernel<EPI_SILU_MUL_F16><<<gg, 256, 0, stream>>>(
            h2, wu_h, nullptr, af, gf, MROWS, FF_, DIM_);
        dim3 gd(DIM_ / 128, MROWS / 128);
        gemm_wmma_kernel<EPI_ADD_F32><<<gd, 256, 0, stream>>>(
            af, wd_h, out, nullptr, x2, MROWS, DIM_, FF_);
    }
}
